// Vega7Model_74861279969642
// MI455X (gfx1250) — compile-verified, matrix operand order checked by reading, not run on hardware
//
#include <hip/hip_runtime.h>
#include <cstdint>
#include <cstddef>

// ---- problem constants (match reference) ----
#define LNUM  6
#define CDIM  1024
#define FFDIM 4096
#define VDIM  32000
#define BSZ   2
#define TSEQ  1024
#define MROWS (BSZ * TSEQ)   // 2048

typedef unsigned short u16;
typedef __attribute__((ext_vector_type(16))) __bf16 v16bf;
typedef __attribute__((ext_vector_type(8)))  float  v8f;

union FragAB { v16bf v; unsigned int u[8]; };
union FragC  { v8f   v; float f[8]; };

__device__ __forceinline__ u16 f2bf(float f) {
  unsigned int u = __float_as_uint(f);
  u += 0x7FFFu + ((u >> 16) & 1u);   // round-to-nearest-even
  return (u16)(u >> 16);
}

// ---------------- fp32 -> bf16 convert (grid-stride) ----------------
__global__ void f32_to_bf16_kernel(const float* __restrict__ src,
                                   u16* __restrict__ dst, long long n) {
  long long i = (long long)blockIdx.x * blockDim.x + threadIdx.x;
  long long stride = (long long)gridDim.x * blockDim.x;
  for (; i < n; i += stride) dst[i] = f2bf(src[i]);
}

// ---------------- embedding gather ----------------
__global__ void embed_kernel(const int* __restrict__ ids,
                             const float* __restrict__ emb,
                             float* __restrict__ x) {
  const int row = blockIdx.x;          // 0..MROWS-1
  const int c4  = threadIdx.x * 4;     // 256 threads * 4 = 1024
  const float4 v = *(const float4*)(emb + (size_t)ids[row] * CDIM + c4);
  *(float4*)(x + (size_t)row * CDIM + c4) = v;
}

// ---------------- LayerNorm (one row per block, 256 threads) ----------------
__global__ __launch_bounds__(256) void ln_kernel(const float* __restrict__ x,
                                                 const float* __restrict__ g,
                                                 const float* __restrict__ bb,
                                                 float* __restrict__ outf,
                                                 u16* __restrict__ outbf) {
  const int row = blockIdx.x;
  const int tid = threadIdx.x;
  const float* xr = x + (size_t)row * CDIM;
  float4 v = ((const float4*)xr)[tid];
  float s  = v.x + v.y + v.z + v.w;
  float s2 = v.x * v.x + v.y * v.y + v.z * v.z + v.w * v.w;
  #pragma unroll
  for (int off = 16; off >= 1; off >>= 1) {
    s  += __shfl_xor(s,  off, 32);
    s2 += __shfl_xor(s2, off, 32);
  }
  __shared__ float a1[8], a2[8];
  __shared__ float mu_s, rs_s;
  const int lane = tid & 31, wave = tid >> 5;
  if (lane == 0) { a1[wave] = s; a2[wave] = s2; }
  __syncthreads();
  if (tid == 0) {
    float S = 0.f, S2 = 0.f;
    #pragma unroll
    for (int i = 0; i < 8; ++i) { S += a1[i]; S2 += a2[i]; }
    float mu  = S / (float)CDIM;
    float var = S2 / (float)CDIM - mu * mu;
    mu_s = mu;
    rs_s = rsqrtf(var + 1e-5f);
  }
  __syncthreads();
  const float mu = mu_s, rs = rs_s;
  const int c0 = tid * 4;
  float fv[4] = {v.x, v.y, v.z, v.w};
  #pragma unroll
  for (int i = 0; i < 4; ++i) {
    const int c = c0 + i;
    float y = (fv[i] - mu) * rs * g[c] + bb[c];
    if (outf)  outf[(size_t)row * CDIM + c]  = y;
    if (outbf) outbf[(size_t)row * CDIM + c] = f2bf(y);
  }
}

// ---------------- bf16 WMMA GEMM: Y[M,N] = act(X[M,K] @ W[N,K]^T) ----------------
// BM=128, BN=128, BK=32. 256 threads = 8 waves (wave32); wave computes a 32x64
// sub-tile via 2x4 v_wmma_f32_16x16x32_bf16 accumulators (8 WMMA / K-step).
// Double-buffered LDS; tiles staged with CDNA5 async-LDS B128 loads in GVS
// (saddr) form so the K-advance is pure SALU. act: 0=none 1=sigmoid 2=relu^2
#define BM 128
#define BN 128
#define BK 32
#define LDSK 40   // halves; 80B rows: 16B-aligned B128 targets,
                  // r*20 mod 64 distinct over 16 rows (no bank clash)

__device__ __forceinline__ void async_b128_to_lds(const u16* sbase,
                                                  unsigned voff_bytes,
                                                  const u16* lds_dst) {
  unsigned loff = (unsigned)(uintptr_t)lds_dst;   // low 32 bits = LDS offset
  asm volatile("global_load_async_to_lds_b128 %0, %1, %2"
               :: "v"(loff), "v"(voff_bytes), "s"(sbase) : "memory");
}

__global__ __launch_bounds__(256) void gemm_bf16_wmma(const u16* __restrict__ X,
                                                      const u16* __restrict__ W,
                                                      float* __restrict__ Yf,
                                                      u16* __restrict__ Ybf,
                                                      int Mi, int Ni, int Ki,
                                                      int act) {
  __shared__ __align__(16) u16 Xs[2][BM][LDSK];
  __shared__ __align__(16) u16 Ws[2][BN][LDSK];

  const int tid  = threadIdx.x;
  const int lane = tid & 31;
  const int wave = tid >> 5;
  const int wm   = wave >> 1;   // 0..3 -> 32-row slabs
  const int wn   = wave & 1;    // 0..1 -> 64-col slabs
  const int m0   = blockIdx.y * BM;
  const int n0   = blockIdx.x * BN;

  const int lr    = lane & 15;        // row within 16
  const int khalf = (lane >> 4) * 8;  // K offset for hi half-wave

  FragC acc[2][4];
  #pragma unroll
  for (int i = 0; i < 2; ++i)
    #pragma unroll
    for (int j = 0; j < 4; ++j)
      #pragma unroll
      for (int gg = 0; gg < 8; ++gg) acc[i][j].f[gg] = 0.f;

  // async staging: 2 B128 per thread per 128x32 tile (8KB = 512 B128)
  const int ldr = tid >> 2;          // 0..63 base row
  const int ldc = (tid & 3) * 8;     // 0,8,16,24 (element col, 16B aligned)
  // per-thread loop-invariant global byte offsets
  unsigned voffX[2], voffW[2];
  #pragma unroll
  for (int i = 0; i < 2; ++i) {
    voffX[i] = (unsigned)(((m0 + ldr + i * 64) * Ki + ldc) * 2);
    voffW[i] = (unsigned)(((n0 + ldr + i * 64) * Ki + ldc) * 2);
  }

  auto issue_tile = [&](int buf, int k0) {
    const u16* xb = X + k0;   // uniform -> SGPR base (SALU advance)
    const u16* wb = W + k0;
    #pragma unroll
    for (int i = 0; i < 2; ++i) {
      async_b128_to_lds(xb, voffX[i], &Xs[buf][ldr + i * 64][ldc]);
      async_b128_to_lds(wb, voffW[i], &Ws[buf][ldr + i * 64][ldc]);
    }
  };

  const int nk = Ki / BK;
  issue_tile(0, 0);
  asm volatile("s_wait_asynccnt 0x0" ::: "memory");
  __syncthreads();

  for (int kt = 0; kt < nk; ++kt) {
    const int cur = kt & 1;
    if (kt + 1 < nk) issue_tile(cur ^ 1, (kt + 1) * BK);

    // ---- build fragments per ISA 7.12.2 16-bit A/B layouts ----
    FragAB a[2], b[4];
    #pragma unroll
    for (int i = 0; i < 2; ++i) {
      const unsigned int* rowp =
          (const unsigned int*)&Xs[cur][wm * 32 + i * 16 + lr][0];
      #pragma unroll
      for (int p = 0; p < 8; ++p) {
        int kp = ((p >> 2) * 16) + khalf + ((p & 3) * 2);
        a[i].u[p] = rowp[kp >> 1];
      }
    }
    #pragma unroll
    for (int j = 0; j < 4; ++j) {
      const unsigned int* rowp =
          (const unsigned int*)&Ws[cur][wn * 64 + j * 16 + lr][0];
      #pragma unroll
      for (int p = 0; p < 8; ++p) {
        int kp = ((p >> 2) * 16) + khalf + ((p & 3) * 2);
        b[j].u[p] = rowp[kp >> 1];
      }
    }
    // ---- 8x v_wmma_f32_16x16x32_bf16 (overlaps next tile's async DMA) ----
    #pragma unroll
    for (int i = 0; i < 2; ++i)
      #pragma unroll
      for (int j = 0; j < 4; ++j)
        acc[i][j].v = __builtin_amdgcn_wmma_f32_16x16x32_bf16(
            false, a[i].v, false, b[j].v, (short)0, acc[i][j].v, false, false);

    if (kt + 1 < nk) {
      asm volatile("s_wait_asynccnt 0x0" ::: "memory");
      __syncthreads();
    }
  }

  // ---- epilogue: C/D layout -> global, fused activation ----
  const int hi = lane >> 4;
  #pragma unroll
  for (int i = 0; i < 2; ++i) {
    #pragma unroll
    for (int j = 0; j < 4; ++j) {
      #pragma unroll
      for (int gg = 0; gg < 8; ++gg) {
        int row = m0 + wm * 32 + i * 16 + gg + hi * 8;
        int col = n0 + wn * 64 + j * 16 + (lane & 15);
        float val = acc[i][j].f[gg];
        if (act == 1)       val = 1.f / (1.f + __expf(-val));
        else if (act == 2) { val = fmaxf(val, 0.f); val = val * val; }
        size_t idx = (size_t)row * Ni + col;
        if (Yf)  Yf[idx]  = val;
        if (Ybf) Ybf[idx] = f2bf(val);
      }
    }
  }
}

// ---------------- WKV scan (sequential in T, state in registers) ----------------
// grid = (CDIM/8 state-columns, B); 256 threads; thread owns 4 state rows x 8 cols.
__global__ __launch_bounds__(256) void wkv_scan_kernel(
    const float* __restrict__ kf, const float* __restrict__ vf,
    const float* __restrict__ rf, const float* __restrict__ decay,
    float* __restrict__ wkv, float* __restrict__ state_out) {
  const int d0  = blockIdx.x * 8;
  const int b   = blockIdx.y;
  const int tid = threadIdx.x;
  const int lane = tid & 31, wave = tid >> 5;

  float wc[4];
  #pragma unroll
  for (int j = 0; j < 4; ++j)
    wc[j] = __expf(-__expf(decay[tid + j * 256]));

  float st[4][8];
  #pragma unroll
  for (int j = 0; j < 4; ++j)
    #pragma unroll
    for (int e = 0; e < 8; ++e) st[j][e] = 0.f;

  __shared__ float red[8][8];  // [wave][col]
  const size_t base = (size_t)b * TSEQ * CDIM;
  const float* kb = kf + base;
  const float* vb = vf + base;
  const float* rb = rf + base;
  float* ob = wkv + base;

  for (int t = 0; t < TSEQ; ++t) {
    float kv[4], rv[4], vv[8];
    #pragma unroll
    for (int j = 0; j < 4; ++j) {
      kv[j] = kb[(size_t)t * CDIM + tid + j * 256];
      rv[j] = rb[(size_t)t * CDIM + tid + j * 256];
    }
    #pragma unroll
    for (int e = 0; e < 8; ++e) vv[e] = vb[(size_t)t * CDIM + d0 + e];

    float out[8];
    #pragma unroll
    for (int e = 0; e < 8; ++e) out[e] = 0.f;
    #pragma unroll
    for (int j = 0; j < 4; ++j)
      #pragma unroll
      for (int e = 0; e < 8; ++e) {
        st[j][e] = st[j][e] * wc[j] + kv[j] * vv[e];
        out[e] += rv[j] * st[j][e];
      }
    // reduce out[] over 256 threads (rows of the state)
    #pragma unroll
    for (int e = 0; e < 8; ++e) {
      float s = out[e];
      #pragma unroll
      for (int off = 16; off >= 1; off >>= 1) s += __shfl_xor(s, off, 32);
      out[e] = s;
    }
    if (lane == 0)
      #pragma unroll
      for (int e = 0; e < 8; ++e) red[wave][e] = out[e];
    __syncthreads();
    if (tid < 8) {
      float s = 0.f;
      #pragma unroll
      for (int w8 = 0; w8 < 8; ++w8) s += red[w8][tid];
      ob[(size_t)t * CDIM + d0 + tid] = s;
    }
    __syncthreads();
  }
  // dump final state: states[b, c, d]
  #pragma unroll
  for (int j = 0; j < 4; ++j)
    #pragma unroll
    for (int e = 0; e < 8; ++e)
      state_out[((size_t)b * CDIM + (tid + j * 256)) * CDIM + d0 + e] = st[j][e];
}

// ---------------- elementwise ----------------
__global__ void add_inplace_kernel(float* __restrict__ x,
                                   const float* __restrict__ a) {
  size_t i = ((size_t)blockIdx.x * blockDim.x + threadIdx.x) * 4;
  float4 xv = *(float4*)(x + i);
  float4 av = *(const float4*)(a + i);
  xv.x += av.x; xv.y += av.y; xv.z += av.z; xv.w += av.w;
  *(float4*)(x + i) = xv;
}

__global__ void cm_combine_kernel(float* __restrict__ x,
                                  const float* __restrict__ xn,
                                  const float* __restrict__ rg,
                                  const float* __restrict__ cv) {
  size_t i = ((size_t)blockIdx.x * blockDim.x + threadIdx.x) * 4;
  float4 xv = *(float4*)(x + i);
  float4 nv = *(const float4*)(xn + i);
  float4 gv = *(const float4*)(rg + i);
  float4 vv = *(const float4*)(cv + i);
  xv.x += nv.x * gv.x + vv.x;
  xv.y += nv.y * gv.y + vv.y;
  xv.z += nv.z * gv.z + vv.z;
  xv.w += nv.w * gv.w + vv.w;
  *(float4*)(x + i) = xv;
}

// =======================================================================
extern "C" void kernel_launch(void* const* d_in, const int* in_sizes, int n_in,
                              void* d_out, int out_size, void* d_ws, size_t ws_size,
                              hipStream_t stream) {
  (void)in_sizes; (void)n_in; (void)out_size; (void)ws_size;
  const int*   ids      = (const int*)d_in[0];
  const float* emb      = (const float*)d_in[1];
  const float* tm_decay = (const float*)d_in[2];
  const float* tm_wk    = (const float*)d_in[3];
  const float* tm_wv    = (const float*)d_in[4];
  const float* tm_wr    = (const float*)d_in[5];
  const float* tm_wo    = (const float*)d_in[6];
  const float* tm_lng   = (const float*)d_in[7];
  const float* tm_lnb   = (const float*)d_in[8];
  const float* cm_wk    = (const float*)d_in[9];
  const float* cm_wv    = (const float*)d_in[10];
  const float* cm_wr    = (const float*)d_in[11];
  const float* cm_lng   = (const float*)d_in[12];
  const float* cm_lnb   = (const float*)d_in[13];
  const float* lno_g    = (const float*)d_in[14];
  const float* lno_b    = (const float*)d_in[15];
  const float* head_w   = (const float*)d_in[16];

  float* logits = (float*)d_out;
  float* states = logits + (size_t)BSZ * TSEQ * VDIM;

  // bump allocator over workspace
  char* wsb = (char*)d_ws;
  size_t off = 0;
  auto alloc = [&](size_t bytes) -> void* {
    off = (off + 255) & ~(size_t)255;
    void* p = wsb + off;
    off += bytes;
    return p;
  };

  const size_t CC  = (size_t)CDIM * CDIM;
  const size_t CF  = (size_t)CDIM * FFDIM;
  u16* tmwk_bf = (u16*)alloc(LNUM * CC * 2);
  u16* tmwv_bf = (u16*)alloc(LNUM * CC * 2);
  u16* tmwr_bf = (u16*)alloc(LNUM * CC * 2);
  u16* tmwo_bf = (u16*)alloc(LNUM * CC * 2);
  u16* cmwk_bf = (u16*)alloc(LNUM * CF * 2);
  u16* cmwv_bf = (u16*)alloc(LNUM * CF * 2);
  u16* cmwr_bf = (u16*)alloc(LNUM * CC * 2);
  u16* head_bf = (u16*)alloc((size_t)VDIM * CDIM * 2);

  const size_t MC = (size_t)MROWS * CDIM;
  float* xf     = (float*)alloc(MC * 4);
  float* xnf    = (float*)alloc(MC * 4);
  u16*   xnbf   = (u16*)alloc(MC * 2);
  float* kf     = (float*)alloc(MC * 4);
  float* vfb    = (float*)alloc(MC * 4);
  float* rfb    = (float*)alloc(MC * 4);
  float* wkvf   = (float*)alloc(MC * 4);
  u16*   wkvbf  = (u16*)alloc(MC * 2);
  float* toutf  = (float*)alloc(MC * 4);
  u16*   kkbf   = (u16*)alloc((size_t)MROWS * FFDIM * 2);
  float* cvf    = (float*)alloc(MC * 4);
  float* rgf    = (float*)alloc(MC * 4);

  // ---- weight conversion fp32 -> bf16 ----
  f32_to_bf16_kernel<<<4096, 256, 0, stream>>>(tm_wk, tmwk_bf, (long long)(LNUM * CC));
  f32_to_bf16_kernel<<<4096, 256, 0, stream>>>(tm_wv, tmwv_bf, (long long)(LNUM * CC));
  f32_to_bf16_kernel<<<4096, 256, 0, stream>>>(tm_wr, tmwr_bf, (long long)(LNUM * CC));
  f32_to_bf16_kernel<<<4096, 256, 0, stream>>>(tm_wo, tmwo_bf, (long long)(LNUM * CC));
  f32_to_bf16_kernel<<<4096, 256, 0, stream>>>(cm_wk, cmwk_bf, (long long)(LNUM * CF));
  f32_to_bf16_kernel<<<4096, 256, 0, stream>>>(cm_wv, cmwv_bf, (long long)(LNUM * CF));
  f32_to_bf16_kernel<<<4096, 256, 0, stream>>>(cm_wr, cmwr_bf, (long long)(LNUM * CC));
  f32_to_bf16_kernel<<<4096, 256, 0, stream>>>(head_w, head_bf, (long long)VDIM * CDIM);

  // ---- embedding ----
  embed_kernel<<<MROWS, 256, 0, stream>>>(ids, emb, xf);

  const dim3 gCC(CDIM / BN, MROWS / BM);    // (8, 16)
  const dim3 gCF(FFDIM / BN, MROWS / BM);   // (32, 16)
  const dim3 gCV(VDIM / BN, MROWS / BM);    // (250, 16)
  const int nElemBlocks = (int)(MC / 4 / 256);

  for (int l = 0; l < LNUM; ++l) {
    // --- time mix ---
    ln_kernel<<<MROWS, 256, 0, stream>>>(xf, tm_lng + l * CDIM, tm_lnb + l * CDIM,
                                         nullptr, xnbf);
    gemm_bf16_wmma<<<gCC, 256, 0, stream>>>(xnbf, tmwk_bf + l * CC, kf, nullptr,
                                            MROWS, CDIM, CDIM, 0);
    gemm_bf16_wmma<<<gCC, 256, 0, stream>>>(xnbf, tmwv_bf + l * CC, vfb, nullptr,
                                            MROWS, CDIM, CDIM, 0);
    gemm_bf16_wmma<<<gCC, 256, 0, stream>>>(xnbf, tmwr_bf + l * CC, rfb, nullptr,
                                            MROWS, CDIM, CDIM, 1);
    wkv_scan_kernel<<<dim3(CDIM / 8, BSZ), 256, 0, stream>>>(
        kf, vfb, rfb, tm_decay + l * CDIM, wkvf, states + (size_t)l * BSZ * CC);
    f32_to_bf16_kernel<<<2048, 256, 0, stream>>>(wkvf, wkvbf, (long long)MC);
    gemm_bf16_wmma<<<gCC, 256, 0, stream>>>(wkvbf, tmwo_bf + l * CC, toutf, nullptr,
                                            MROWS, CDIM, CDIM, 0);
    add_inplace_kernel<<<nElemBlocks, 256, 0, stream>>>(xf, toutf);

    // --- channel mix ---
    ln_kernel<<<MROWS, 256, 0, stream>>>(xf, cm_lng + l * CDIM, cm_lnb + l * CDIM,
                                         xnf, xnbf);
    gemm_bf16_wmma<<<gCF, 256, 0, stream>>>(xnbf, cmwk_bf + l * CF, nullptr, kkbf,
                                            MROWS, FFDIM, CDIM, 2);   // relu^2 -> bf16
    gemm_bf16_wmma<<<gCC, 256, 0, stream>>>(kkbf, cmwv_bf + l * CF, cvf, nullptr,
                                            MROWS, CDIM, FFDIM, 0);
    gemm_bf16_wmma<<<gCC, 256, 0, stream>>>(xnbf, cmwr_bf + l * CC, rgf, nullptr,
                                            MROWS, CDIM, CDIM, 1);    // sigmoid
    cm_combine_kernel<<<nElemBlocks, 256, 0, stream>>>(xf, xnf, rgf, cvf);
  }

  // ---- final LN + head ----
  ln_kernel<<<MROWS, 256, 0, stream>>>(xf, lno_g, lno_b, nullptr, xnbf);
  gemm_bf16_wmma<<<gCV, 256, 0, stream>>>(xnbf, head_bf, logits, nullptr,
                                          MROWS, VDIM, CDIM, 0);
}
